// Encoder_89550068122034
// MI455X (gfx1250) — compile-verified, matrix operand order checked by reading, not run on hardware
//
#include <hip/hip_runtime.h>

// Problem dims
#define T_LEN   512
#define BATCH   64
#define IN_DIMS 1024
#define HID     256
#define G3      768          // 3*H
#define NTOT    1536         // 2*G3 (both directions packed)
#define MROWS   32768        // T*B

typedef __attribute__((ext_vector_type(16))) __bf16 v16bf;
typedef __attribute__((ext_vector_type(8)))  __bf16 v8bf;
typedef __attribute__((ext_vector_type(8)))  float  v8f;

__device__ __forceinline__ unsigned short f2bf(float f) {
  unsigned u = __float_as_uint(f);
  u += 0x7FFF + ((u >> 16) & 1);          // round-to-nearest-even
  return (unsigned short)(u >> 16);
}
__device__ __forceinline__ float bf2f(unsigned short b) {
  return __uint_as_float(((unsigned)b) << 16);
}
__device__ __forceinline__ float sigm(float x) { return 1.0f / (1.0f + __expf(-x)); }

// k-within-tile for A/B bf16 fragments (ISA 16-bit A 16x32 layout):
// lanes 0-15: e0..7 -> K=0..7, e8..15 -> K=16..23 ; lanes 16-31: +8
__device__ __forceinline__ int k_in_tile(int lane, int e) {
  return (e & 7) + ((e >> 3) << 4) + ((lane >> 4) << 3);
}

// ---------------- packing kernels ----------------
// Pack src fp32 (MROWS x IN_DIMS) -> A-fragment bf16 [mTile][kTile(32)][lane][16]
__global__ void k_pack_a_src(const float* __restrict__ src, unsigned short* __restrict__ out) {
  const long total = (long)(MROWS / 16) * (IN_DIMS / 32) * 512;
  for (long idx = (long)blockIdx.x * blockDim.x + threadIdx.x; idx < total;
       idx += (long)gridDim.x * blockDim.x) {
    int  e     = (int)(idx & 15);
    int  lane  = (int)((idx >> 4) & 31);
    int  kTile = (int)((idx >> 9) & 31);
    long mTile = idx >> 14;
    long m = (mTile << 4) + (lane & 15);
    int  k = (kTile << 5) + k_in_tile(lane, e);
    out[idx] = f2bf(src[m * IN_DIMS + k]);
  }
}

// Pack weight W (N x K row-major, fp32) as WMMA B fragments: B[k][n] = W[n][k]
// layout [nTile][kTile][lane][16]
__global__ void k_pack_b_w(const float* __restrict__ W, int N, int K,
                           unsigned short* __restrict__ out) {
  const int  kTiles = K >> 5;
  const long total  = (long)(N >> 4) * kTiles * 512;
  for (long idx = (long)blockIdx.x * blockDim.x + threadIdx.x; idx < total;
       idx += (long)gridDim.x * blockDim.x) {
    int  e     = (int)(idx & 15);
    int  lane  = (int)((idx >> 4) & 31);
    long rest  = idx >> 9;
    int  kTile = (int)(rest % kTiles);
    int  nTile = (int)(rest / kTiles);
    int  n = (nTile << 4) + (lane & 15);
    int  k = (kTile << 5) + k_in_tile(lane, e);
    out[idx] = f2bf(W[(long)n * K + k]);
  }
}

// ---------------- input projection GEMM: gx = src @ [Wih_f|Wih_b]^T + bih ----------------
// M=32768, K=1024, N=1536. Block tile 128x128, 8 waves (2M x 4N), wave = 4x2 tiles.
__global__ void k_gemm_gx(const unsigned short* __restrict__ aPack,
                          const unsigned short* __restrict__ bPack,
                          const float* __restrict__ bih_f,
                          const float* __restrict__ bih_b,
                          unsigned short* __restrict__ gx) {
  const int lane = threadIdx.x & 31;
  const int w    = threadIdx.x >> 5;
  const int wm   = w & 1, wn = w >> 1;
  const int mT0  = blockIdx.x * 8 + wm * 4;   // 4 m-tiles
  const int nT0  = blockIdx.y * 8 + wn * 2;   // 2 n-tiles
  const v16bf* A  = (const v16bf*)aPack;
  const v16bf* Bv = (const v16bf*)bPack;
  v8f acc[4][2] = {};
#pragma unroll 4
  for (int k = 0; k < 32; ++k) {
    if (k + 1 < 32) {
      // near-cache (WGP scope) prefetch of next k-step fragments
      __builtin_prefetch((const void*)&A[((long)mT0 * 32 + k + 1) * 32 + lane], 0, 3);
      __builtin_prefetch((const void*)&Bv[((long)nT0 * 32 + k + 1) * 32 + lane], 0, 3);
    }
    v16bf a0 = A[((long)(mT0 + 0) * 32 + k) * 32 + lane];
    v16bf a1 = A[((long)(mT0 + 1) * 32 + k) * 32 + lane];
    v16bf a2 = A[((long)(mT0 + 2) * 32 + k) * 32 + lane];
    v16bf a3 = A[((long)(mT0 + 3) * 32 + k) * 32 + lane];
    v16bf b0 = Bv[((long)(nT0 + 0) * 32 + k) * 32 + lane];
    v16bf b1 = Bv[((long)(nT0 + 1) * 32 + k) * 32 + lane];
    acc[0][0] = __builtin_amdgcn_wmma_f32_16x16x32_bf16(false, a0, false, b0, (short)0, acc[0][0], false, false);
    acc[1][0] = __builtin_amdgcn_wmma_f32_16x16x32_bf16(false, a1, false, b0, (short)0, acc[1][0], false, false);
    acc[2][0] = __builtin_amdgcn_wmma_f32_16x16x32_bf16(false, a2, false, b0, (short)0, acc[2][0], false, false);
    acc[3][0] = __builtin_amdgcn_wmma_f32_16x16x32_bf16(false, a3, false, b0, (short)0, acc[3][0], false, false);
    acc[0][1] = __builtin_amdgcn_wmma_f32_16x16x32_bf16(false, a0, false, b1, (short)0, acc[0][1], false, false);
    acc[1][1] = __builtin_amdgcn_wmma_f32_16x16x32_bf16(false, a1, false, b1, (short)0, acc[1][1], false, false);
    acc[2][1] = __builtin_amdgcn_wmma_f32_16x16x32_bf16(false, a2, false, b1, (short)0, acc[2][1], false, false);
    acc[3][1] = __builtin_amdgcn_wmma_f32_16x16x32_bf16(false, a3, false, b1, (short)0, acc[3][1], false, false);
  }
  const int half = lane >> 4;
#pragma unroll
  for (int j = 0; j < 2; ++j) {
    int nG = (nT0 + j) * 16 + (lane & 15);
    float bias = (nG < G3) ? bih_f[nG] : bih_b[nG - G3];
#pragma unroll
    for (int i = 0; i < 4; ++i) {
#pragma unroll
      for (int r = 0; r < 8; ++r) {
        long m = (long)(mT0 + i) * 16 + r + 8 * half;
        gx[m * NTOT + nG] = f2bf(acc[i][j][r] + bias);
      }
    }
  }
}

// ---------------- recurrent scan: one 1024-thread workgroup per direction ----------------
// LDS: h32 (64x256 f32, 64KB) | hbf (64x256 bf16, 32KB) | gh (64x768 f32, 192KB) = 288KB
__global__ void k_gru_scan(const unsigned short* __restrict__ gx,
                           const unsigned short* __restrict__ whhPack_f,
                           const unsigned short* __restrict__ whhPack_b,
                           const float* __restrict__ bhh_f,
                           const float* __restrict__ bhh_b,
                           unsigned short* __restrict__ outPack,
                           float* __restrict__ hfinal) {
  extern __shared__ char smem[];
  float*          h32 = (float*)smem;                                   // 64*256
  unsigned short* hbf = (unsigned short*)(smem + 64 * 256 * 4);         // 64*256
  float*          gh  = (float*)(smem + 64 * 256 * 4 + 64 * 256 * 2);   // 64*768

  const int dir = blockIdx.x;
  const v16bf* Wv  = (const v16bf*)(dir ? whhPack_b : whhPack_f);
  const float* bhh = dir ? bhh_b : bhh_f;

  const int tid  = threadIdx.x;
  const int lane = tid & 31;
  const int w    = tid >> 5;         // 0..31 waves
  const int m0   = (w & 3) * 16;     // 4 waves in M
  const int nw   = w >> 2;           // 8 wave-groups in N, 6 n-tiles each

  for (int i = tid; i < 64 * 256; i += 1024) { h32[i] = 0.0f; hbf[i] = 0; }
  __syncthreads();

  const int bb = tid >> 4;           // phase-2 row (0..63)
  const int j0 = (tid & 15) * 16;    // phase-2 16-col slice
  const int mrow   = m0 + (lane & 15);
  const int khalf8 = (lane >> 4) * 8;

  for (int t = 0; t < T_LEN; ++t) {
    // ---- phase 1: gh = h_bf16 @ Whh^T via WMMA ----
    v8f acc[6] = {};
#pragma unroll
    for (int k = 0; k < 8; ++k) {
      const v8bf lo = *(const v8bf*)(hbf + mrow * 256 + k * 32 + khalf8);
      const v8bf hi = *(const v8bf*)(hbf + mrow * 256 + k * 32 + 16 + khalf8);
      v16bf a = __builtin_shufflevector(lo, hi, 0, 1, 2, 3, 4, 5, 6, 7,
                                        8, 9, 10, 11, 12, 13, 14, 15);
#pragma unroll
      for (int i = 0; i < 6; ++i) {
        v16bf b = Wv[((long)(nw * 6 + i) * 8 + k) * 32 + lane];
        acc[i] = __builtin_amdgcn_wmma_f32_16x16x32_bf16(false, a, false, b, (short)0, acc[i], false, false);
      }
    }
    {
      const int rowBase = m0 + 8 * (lane >> 4);
#pragma unroll
      for (int i = 0; i < 6; ++i) {
        int nG = (nw * 6 + i) * 16 + (lane & 15);
#pragma unroll
        for (int r = 0; r < 8; ++r) gh[(rowBase + r) * G3 + nG] = acc[i][r];
      }
    }
    __syncthreads();

    // ---- phase 2: gates + state update + packed output write ----
    const int  trow   = dir ? (T_LEN - 1 - t) : t;
    const long gxbase = ((long)trow * BATCH + bb) * NTOT + (long)dir * G3;
    const long mOut   = (long)trow * BATCH + bb;
    const long mTile  = mOut >> 4;
    const int  mIn    = (int)(mOut & 15);
#pragma unroll 4
    for (int jj = 0; jj < 16; ++jj) {
      int   j  = j0 + jj;
      float gr = bf2f(gx[gxbase + j]);
      float gz = bf2f(gx[gxbase + 256 + j]);
      float gn = bf2f(gx[gxbase + 512 + j]);
      float hr = gh[bb * G3 + j]       + bhh[j];
      float hz = gh[bb * G3 + 256 + j] + bhh[256 + j];
      float hn = gh[bb * G3 + 512 + j] + bhh[512 + j];
      float r  = sigm(gr + hr);
      float z  = sigm(gz + hz);
      float n  = tanhf(gn + r * hn);
      float hnew = (1.0f - z) * n + z * h32[bb * 256 + j];
      h32[bb * 256 + j] = hnew;
      unsigned short hb = f2bf(hnew);
      hbf[bb * 256 + j] = hb;
      // write directly into A-fragment layout for the output GEMM (K=512, kTiles=16)
      int kG = dir * 256 + j;
      int kT = kG >> 5, kk = kG & 31;
      int laneIdx = mIn + (((kk >> 3) & 1) << 4);
      int e       = (kk & 7) | (((kk >> 4) & 1) << 3);
      outPack[((mTile * 16 + kT) * 32 + laneIdx) * 16 + e] = hb;
    }
    __syncthreads();
  }

  // final hidden state (own elements, no sync needed)
  for (int jj = 0; jj < 16; ++jj) {
    int j = j0 + jj;
    hfinal[(long)dir * BATCH * HID + bb * 256 + j] = h32[bb * 256 + j];
  }
}

// ---------------- output projection GEMM: out = concat(h_f,h_b) @ Wout^T + bout ----------------
// M=32768, K=512, N=256. Block = 4 m-tiles x all 16 n-tiles; 8 waves = 4M x 2N.
__global__ void k_gemm_out(const unsigned short* __restrict__ aPack,
                           const unsigned short* __restrict__ bPack,
                           const float* __restrict__ bout,
                           float* __restrict__ out) {
  const int lane = threadIdx.x & 31;
  const int w    = threadIdx.x >> 5;
  const int wm   = w & 3, wn = w >> 2;
  const long mT  = (long)blockIdx.x * 4 + wm;
  const int  nT0 = wn * 8;
  const v16bf* A  = (const v16bf*)aPack;
  const v16bf* Bv = (const v16bf*)bPack;
  v8f acc[8] = {};
#pragma unroll 2
  for (int k = 0; k < 16; ++k) {
    v16bf a = A[(mT * 16 + k) * 32 + lane];
#pragma unroll
    for (int i = 0; i < 8; ++i) {
      v16bf b = Bv[((long)(nT0 + i) * 16 + k) * 32 + lane];
      acc[i] = __builtin_amdgcn_wmma_f32_16x16x32_bf16(false, a, false, b, (short)0, acc[i], false, false);
    }
  }
  const long m0 = mT * 16 + 8 * (lane >> 4);
#pragma unroll
  for (int i = 0; i < 8; ++i) {
    int nG = (nT0 + i) * 16 + (lane & 15);
    float bias = bout[nG];
#pragma unroll
    for (int r = 0; r < 8; ++r) out[(m0 + r) * HID + nG] = acc[i][r] + bias;
  }
}

// ---------------- hidden projection: tanh(concat(h_f,h_b) @ Whid^T + bhid) ----------------
__global__ void k_hidden(const float* __restrict__ hfinal,
                         const float* __restrict__ Whid,
                         const float* __restrict__ bhid,
                         float* __restrict__ out) {
  int o = blockIdx.x * blockDim.x + threadIdx.x;
  if (o >= BATCH * HID) return;
  int b = o >> 8, oc = o & 255;
  const float* hf = hfinal + (long)b * 256;
  const float* hb = hfinal + (long)BATCH * HID + (long)b * 256;
  const float* wr = Whid + (long)oc * 512;
  float s = bhid[oc];
  for (int k = 0; k < 256; ++k) s += hf[k] * wr[k];
  for (int k = 0; k < 256; ++k) s += hb[k] * wr[256 + k];
  out[o] = tanhf(s);
}

// ---------------- host launcher ----------------
extern "C" void kernel_launch(void* const* d_in, const int* in_sizes, int n_in,
                              void* d_out, int out_size, void* d_ws, size_t ws_size,
                              hipStream_t stream) {
  const float* src   = (const float*)d_in[0];
  const float* Wih_f = (const float*)d_in[1];
  const float* Whh_f = (const float*)d_in[2];
  const float* bih_f = (const float*)d_in[3];
  const float* bhh_f = (const float*)d_in[4];
  const float* Wih_b = (const float*)d_in[5];
  const float* Whh_b = (const float*)d_in[6];
  const float* bih_b = (const float*)d_in[7];
  const float* bhh_b = (const float*)d_in[8];
  const float* Wout  = (const float*)d_in[9];
  const float* bout  = (const float*)d_in[10];
  const float* Whid  = (const float*)d_in[11];
  const float* bhid  = (const float*)d_in[12];
  float* out = (float*)d_out;

  // workspace layout (ushort elements unless noted)
  char* ws = (char*)d_ws;
  unsigned short* srcPack  = (unsigned short*)ws;                    // 2048*32*512
  ws += (size_t)MROWS * IN_DIMS * 2;                                 // 64 MB
  unsigned short* wihPack  = (unsigned short*)ws;                    // 96*32*512
  ws += (size_t)96 * 32 * 512 * 2;
  unsigned short* whhPackF = (unsigned short*)ws;                    // 48*8*512
  ws += (size_t)48 * 8 * 512 * 2;
  unsigned short* whhPackB = (unsigned short*)ws;
  ws += (size_t)48 * 8 * 512 * 2;
  unsigned short* woutPack = (unsigned short*)ws;                    // 16*16*512
  ws += (size_t)16 * 16 * 512 * 2;
  unsigned short* gxBuf    = (unsigned short*)ws;                    // 32768*1536
  ws += (size_t)MROWS * NTOT * 2;
  unsigned short* outPack  = (unsigned short*)ws;                    // 2048*16*512
  ws += (size_t)MROWS * 512 * 2;
  float* hfinal            = (float*)ws;                             // 2*64*256 f32

  // 1) pack inputs/weights to bf16 WMMA fragment layouts
  k_pack_a_src<<<2048, 256, 0, stream>>>(src, srcPack);
  k_pack_b_w<<<512, 256, 0, stream>>>(Wih_f, G3, IN_DIMS, wihPack);
  k_pack_b_w<<<512, 256, 0, stream>>>(Wih_b, G3, IN_DIMS, wihPack + (size_t)48 * 32 * 512);
  k_pack_b_w<<<256, 256, 0, stream>>>(Whh_f, G3, HID, whhPackF);
  k_pack_b_w<<<256, 256, 0, stream>>>(Whh_b, G3, HID, whhPackB);
  k_pack_b_w<<<128, 256, 0, stream>>>(Wout, HID, 2 * HID, woutPack);

  // 2) input projection GEMM (both directions fused, N=1536)
  k_gemm_gx<<<dim3(MROWS / 128, NTOT / 128), 256, 0, stream>>>(
      srcPack, wihPack, bih_f, bih_b, gxBuf);

  // 3) recurrent scan: one WGP per direction, 288 KB dynamic LDS
  k_gru_scan<<<2, 1024, 64 * 256 * 4 + 64 * 256 * 2 + 64 * G3 * 4, stream>>>(
      gxBuf, whhPackF, whhPackB, bhh_f, bhh_b, outPack, hfinal);

  // 4) output projection GEMM -> d_out[0 .. T*B*H)
  k_gemm_out<<<MROWS / 64, 256, 0, stream>>>(outPack, woutPack, bout, out);

  // 5) hidden projection -> d_out[T*B*H ..)
  k_hidden<<<BATCH, 256, 0, stream>>>(hfinal, Whid, bhid, out + (size_t)MROWS * HID);
}
